// SpatioTemporalAttentionGAT_Two_5600637354701
// MI455X (gfx1250) — matrix operand order repaired and checked by reading, compiled, not run on hardware
//
#include <hip/hip_runtime.h>

// ---------------- problem dimensions (match reference) ----------------
#define B_      8
#define T_      8
#define N_      64
#define H_      64
#define HEADS_  4
#define NECK_   256
#define E_      1024
#define EP_     1088        // E + N self loops
#define BTN_    4096        // B*T*N images

// ---------------- CDNA5 WMMA / TDM types ----------------
typedef __attribute__((ext_vector_type(16))) __bf16 v16bf;
typedef __attribute__((ext_vector_type(8)))  float  v8f;
typedef unsigned int u32;
typedef __attribute__((ext_vector_type(4))) u32 u32x4;
typedef __attribute__((ext_vector_type(4))) int i32x4;
typedef __attribute__((ext_vector_type(8))) int i32x8;

// A fragment: 16x32 (MxK) bf16, wave32, contiguous k order used by this code:
//  lanes 0-15: M=lane, VGPR v holds K = 2*v (+1)
//  lanes 16-31: same M, K offset +16
__device__ __forceinline__ v16bf load_a16(const float* __restrict__ A, int stride_m,
                                          int k0, int lane) {
  int m   = lane & 15;
  int khi = (lane >> 4) << 4;
  const float* row = A + m * stride_m + k0 + khi;
  v16bf a;
#pragma unroll
  for (int v = 0; v < 8; ++v) {
    int koff = v << 1;   // 2*v
    a[2 * v]     = (__bf16)row[koff];
    a[2 * v + 1] = (__bf16)row[koff + 1];
  }
  return a;
}

// B fragment: 32x16 (KxN) bf16. lanes 0-15 hold K=0..15 for column N=lane,
// lanes 16-31 hold K=16..31. Weight stored [N,K] row-major ("out,in"):
__device__ __forceinline__ v16bf load_b_nk(const float* __restrict__ W, int stride_n,
                                           int k0, int lane) {
  int n  = lane & 15;
  int kb = (lane >> 4) << 4;
  const float* row = W + n * stride_n + k0 + kb;
  v16bf b;
#pragma unroll
  for (int i = 0; i < 16; ++i) b[i] = (__bf16)row[i];
  return b;
}

// Weight stored [K,N] row-major:
__device__ __forceinline__ v16bf load_b_kn(const float* __restrict__ W, int ldn,
                                           int k0, int lane) {
  int n  = lane & 15;
  int kb = (lane >> 4) << 4;
  const float* p = W + (long)(k0 + kb) * ldn + n;
  v16bf b;
#pragma unroll
  for (int i = 0; i < 16; ++i) b[i] = (__bf16)p[i * ldn];
  return b;
}

// ---------------- encoder: 3x3 stride-2 pad-1 conv + ReLU ----------------
__global__ void enc_conv_kernel(const float* __restrict__ in, const float* __restrict__ w,
                                const float* __restrict__ bias, float* __restrict__ out,
                                int Cin, int Hin, int Cout) {
  int Hout = Hin >> 1;
  int total = BTN_ * Cout * Hout * Hout;
  int idx = blockIdx.x * blockDim.x + threadIdx.x;
  if (idx >= total) return;
  int x = idx % Hout; int r = idx / Hout;
  int y = r % Hout;   r /= Hout;
  int oc = r % Cout;  int g = r / Cout;
  const float* inb = in + (long)g * Cin * Hin * Hin;
  float acc = bias[oc];
  for (int c = 0; c < Cin; ++c) {
    const float* ip = inb + c * Hin * Hin;
    const float* wp = w + ((oc * Cin + c) * 9);
#pragma unroll
    for (int ky = 0; ky < 3; ++ky) {
      int iy = 2 * y - 1 + ky;
      if (iy < 0 || iy >= Hin) continue;
#pragma unroll
      for (int kx = 0; kx < 3; ++kx) {
        int ix = 2 * x - 1 + kx;
        if (ix < 0 || ix >= Hin) continue;
        acc += wp[ky * 3 + kx] * ip[iy * Hin + ix];
      }
    }
  }
  out[idx] = fmaxf(acc, 0.0f);
}

// ---------------- per-node QKV projection: 64 GEMMs [64x256]@[256x768] ----------------
__global__ __launch_bounds__(256)
void qkv_wmma_kernel(const float* __restrict__ xt,   // [4096,256], img=(b*8+t)*64+n
                     const float* __restrict__ Wq,   // [64,768,256]  (n, f, e)
                     const float* __restrict__ bq,   // [64,768]
                     float* __restrict__ qkv) {      // [4096,768]
  int tile = blockIdx.x * (blockDim.x >> 5) + (threadIdx.x >> 5);  // 64*4*48 tiles
  int lane = threadIdx.x & 31;
  int node = tile / (4 * 48);
  int rem  = tile % (4 * 48);
  int mt = rem / 48, nt = rem % 48;
  const float* A = xt + ((mt * 16) * N_ + node) * NECK_;
  const int lda = N_ * NECK_;
  const float* W = Wq + ((long)node * 768 + nt * 16) * NECK_;
  v8f c = {};
#pragma unroll
  for (int k0 = 0; k0 < NECK_; k0 += 32) {
    v16bf a = load_a16(A, lda, k0, lane);
    v16bf b = load_b_nk(W, NECK_, k0, lane);
    c = __builtin_amdgcn_wmma_f32_16x16x32_bf16(false, a, false, b, (short)0, c, false, false);
  }
  int nloc = lane & 15;
  int mb   = (lane >> 4) << 3;
  int fcol = nt * 16 + nloc;
  float bias = bq[node * 768 + fcol];
  float* D = qkv + ((mt * 16 + mb) * N_ + node) * 768 + fcol;
#pragma unroll
  for (int r = 0; r < 8; ++r) D[r * N_ * 768] = c[r] + bias;
}

// ---------------- per-(b,t,n,h) attention over the B axis (seq=8) ----------------
__global__ void attention_kernel(const float* __restrict__ qkv, float* __restrict__ o) {
  int idx = blockIdx.x * blockDim.x + threadIdx.x;   // (b,t,n,h) = 16384
  if (idx >= B_ * T_ * N_ * HEADS_) return;
  int hh = idx & 3; int r = idx >> 2;
  int n = r & 63;   r >>= 6;
  int t = r & 7;    int b = r >> 3;
  const float* qp = qkv + ((b * 8 + t) * N_ + n) * 768 + hh * 64;
  float s[8]; float mx = -1e30f;
#pragma unroll
  for (int ss = 0; ss < 8; ++ss) {
    const float* kp = qkv + ((ss * 8 + t) * N_ + n) * 768 + 256 + hh * 64;
    float acc = 0.f;
    for (int d = 0; d < 64; ++d) acc += qp[d] * kp[d];
    acc *= 0.125f;                     // d^-0.5, d=64
    s[ss] = acc; mx = fmaxf(mx, acc);
  }
  float den = 0.f;
#pragma unroll
  for (int ss = 0; ss < 8; ++ss) { s[ss] = __expf(s[ss] - mx); den += s[ss]; }
  float inv = 1.f / den;
  float* op = o + ((b * 8 + t) * N_ + n) * NECK_ + hh * 64;
  for (int d = 0; d < 64; ++d) {
    float acc = 0.f;
#pragma unroll
    for (int ss = 0; ss < 8; ++ss)
      acc += s[ss] * qkv[((ss * 8 + t) * N_ + n) * 768 + 512 + hh * 64 + d];
    op[d] = acc * inv;
  }
}

// ---- per-node output projection; stores straight into [B,N,T*NECK] GAT layout ----
__global__ __launch_bounds__(256)
void attnout_wmma_kernel(const float* __restrict__ oa,  // [4096,256]
                         const float* __restrict__ Wo,  // [64,256,256]
                         const float* __restrict__ bo,  // [64,256]
                         float* __restrict__ af) {      // [B,N,2048]
  int tile = blockIdx.x * (blockDim.x >> 5) + (threadIdx.x >> 5);  // 64*4*16
  int lane = threadIdx.x & 31;
  int node = tile / 64;
  int rem  = tile % 64;
  int mt = rem / 16, nt = rem % 16;
  const float* A = oa + ((mt * 16) * N_ + node) * NECK_;
  const float* W = Wo + ((long)node * 256 + nt * 16) * NECK_;
  v8f c = {};
#pragma unroll
  for (int k0 = 0; k0 < NECK_; k0 += 32) {
    v16bf a = load_a16(A, N_ * NECK_, k0, lane);
    v16bf b = load_b_nk(W, NECK_, k0, lane);
    c = __builtin_amdgcn_wmma_f32_16x16x32_bf16(false, a, false, b, (short)0, c, false, false);
  }
  int nloc = lane & 15;
  int mb   = (lane >> 4) << 3;
  int fcol = nt * 16 + nloc;
  float bias = bo[node * 256 + fcol];
#pragma unroll
  for (int r = 0; r < 8; ++r) {
    int m = mt * 16 + mb + r;              // b*8+t
    int b = m >> 3, t = m & 7;
    af[((long)b * N_ + node) * 2048 + t * 256 + fcol] = c[r] + bias;
  }
}

// ---------------- GAT linear: [512 x 2048] @ [2048 x 1024] ----------------
// Block = 8 waves sharing one 16-row A panel staged into LDS by the Tensor
// Data Mover (TDM); each wave owns one 16-col N tile. A is read 8x from LDS
// instead of 8x from global; TDM DMA overlaps with the B-operand streaming.
#define GAT_KP 256            // K panel staged per TDM transfer (16x256 f32 = 16KB)

__global__ __launch_bounds__(256)
void gat_wmma_kernel(const float* __restrict__ af,  // [512,2048]
                     const float* __restrict__ W,   // [2048,1024]
                     float* __restrict__ h) {       // [512,1024]
  __shared__ float As[16 * GAT_KP];
  int wave = threadIdx.x >> 5;
  int lane = threadIdx.x & 31;
  int mt  = blockIdx.x >> 3;          // 32 M tiles
  int ntg = blockIdx.x & 7;           // 8 groups of 8 N tiles
  int nt  = ntg * 8 + wave;
  const float* Abase = af + (long)(mt * 16) * 2048;
  const float* Wp    = W + nt * 16;
  v8f c = {};
  for (int kp = 0; kp < 2048; kp += GAT_KP) {
    __syncthreads();                  // previous panel fully consumed
#if __has_builtin(__builtin_amdgcn_tensor_load_to_lds) && __has_builtin(__builtin_amdgcn_s_wait_tensorcnt)
    if (wave == 0) {
      // Tensor DMA descriptor (ISA 8.3/8.4): 2D tile, 4-byte elements,
      // tile 256x16, row stride 2048 elements, dest = As in LDS.
      unsigned long long ga = (unsigned long long)(const void*)(Abase + kp);
      u32 lds = (u32)(unsigned long long)(void*)&As[0];
      u32x4 g0;
      g0[0] = 1u;                                           // count=1, user D#
      g0[1] = lds;                                          // lds_addr
      g0[2] = (u32)ga;                                      // global_addr[31:0]
      g0[3] = (u32)((ga >> 32) & 0x01FFFFFFu) | 0x80000000u;// addr[56:32] | type=2
      i32x8 g1;
      g1[0] = 0x00020000;               // workgroup_mask=0, data_size=4B
      g1[1] = (int)(2048u << 16);       // tensor_dim0[15:0]  @ bits 63:48
      g1[2] = (int)(16u   << 16);       // tensor_dim1[15:0]  @ bits 95:80
      g1[3] = (int)((u32)GAT_KP << 16); // tile_dim0          @ bits 127:112
      g1[4] = 16;                       // tile_dim1          @ bits 143:128
      g1[5] = 2048;                     // tensor_dim0_stride[31:0]
      g1[6] = 0;
      g1[7] = 0;
      i32x4 g2 = {0, 0, 0, 0};
      i32x4 g3 = {0, 0, 0, 0};
      i32x8 g4 = {0, 0, 0, 0, 0, 0, 0, 0};   // 6-arg toolchain form
      __builtin_amdgcn_tensor_load_to_lds(g0, g1, g2, g3, g4, 0);
      __builtin_amdgcn_s_wait_tensorcnt((short)0);
    }
#else
    // cooperative fallback staging
    for (int i = threadIdx.x; i < 16 * GAT_KP; i += 256)
      As[i] = Abase[(i >> 8) * 2048 + kp + (i & (GAT_KP - 1))];
#endif
    __syncthreads();                  // panel visible to all 8 waves
    // prefetch next B panel into GL2 while we crunch this one
    if (kp + GAT_KP < 2048)
      __builtin_prefetch((const void*)(Wp + (long)(kp + GAT_KP) * 1024 + lane), 0, 0);
#pragma unroll
    for (int k0 = 0; k0 < GAT_KP; k0 += 32) {
      v16bf a = load_a16(&As[0], GAT_KP, k0, lane);     // ds reads
      v16bf b = load_b_kn(Wp, 1024, kp + k0, lane);     // global streams
      c = __builtin_amdgcn_wmma_f32_16x16x32_bf16(false, a, false, b, (short)0, c, false, false);
    }
  }
  int nloc = lane & 15;
  int mb   = (lane >> 4) << 3;
#pragma unroll
  for (int r = 0; r < 8; ++r)
    h[(long)(mt * 16 + mb + r) * 1024 + nt * 16 + nloc] = c[r];
}

// ---------------- GAT attention coefficients ----------------
__global__ void gat_attcoef_kernel(const float* __restrict__ h, const float* __restrict__ asrc,
                                   const float* __restrict__ adst,
                                   float* __restrict__ a_s, float* __restrict__ a_d) {
  int idx = blockIdx.x * blockDim.x + threadIdx.x;   // (b,n,hh) = 2048
  if (idx >= B_ * N_ * HEADS_) return;
  int hh = idx & 3; int bn = idx >> 2;
  const float* hp = h + (long)bn * 1024 + hh * 256;
  float s = 0.f, d = 0.f;
  for (int i = 0; i < 256; ++i) { s += hp[i] * asrc[hh * 256 + i]; d += hp[i] * adst[hh * 256 + i]; }
  a_s[idx] = s; a_d[idx] = d;
}

// deterministic per-destination segment max + denominator (incl. self loops)
__global__ void gat_maxden_kernel(const int* __restrict__ ei, const float* __restrict__ a_s,
                                  const float* __restrict__ a_d,
                                  float* __restrict__ mx, float* __restrict__ den) {
  int idx = blockIdx.x * blockDim.x + threadIdx.x;   // (b,n,hh) = 2048
  if (idx >= B_ * N_ * HEADS_) return;
  int hh = idx & 3; int n = (idx >> 2) & 63; int b = idx >> 8;
  float ad = a_d[idx];
  float m = -1e30f;
  for (int e = 0; e < EP_; ++e) {
    int dst = (e < E_) ? ei[E_ + e] : (e - E_);
    if (dst != n) continue;
    int src = (e < E_) ? ei[e] : (e - E_);
    float v = a_s[(b * N_ + src) * 4 + hh] + ad;
    v = v > 0.f ? v : 0.2f * v;
    m = fmaxf(m, v);
  }
  float s = 0.f;
  for (int e = 0; e < EP_; ++e) {
    int dst = (e < E_) ? ei[E_ + e] : (e - E_);
    if (dst != n) continue;
    int src = (e < E_) ? ei[e] : (e - E_);
    float v = a_s[(b * N_ + src) * 4 + hh] + ad;
    v = v > 0.f ? v : 0.2f * v;
    s += __expf(v - m);
  }
  mx[idx] = m; den[idx] = s;
}

// aggregate messages per (b, dst) — block per (b,n), thread owns (h, d=tid)
__global__ __launch_bounds__(256)
void gat_agg_kernel(const int* __restrict__ ei, const float* __restrict__ h,
                    const float* __restrict__ a_s, const float* __restrict__ a_d,
                    const float* __restrict__ mx, const float* __restrict__ den,
                    float* __restrict__ agg) {
  int bn = blockIdx.x;              // 512
  int b = bn >> 6, n = bn & 63;
  int tid = threadIdx.x;            // d index within each head
  float ad[4], ml[4], dl[4], acc[4] = {0.f, 0.f, 0.f, 0.f};
#pragma unroll
  for (int j = 0; j < 4; ++j) {
    ad[j] = a_d[bn * 4 + j]; ml[j] = mx[bn * 4 + j]; dl[j] = den[bn * 4 + j];
  }
  for (int e = 0; e < EP_; ++e) {
    int dst = (e < E_) ? ei[E_ + e] : (e - E_);
    if (dst != n) continue;
    int src = (e < E_) ? ei[e] : (e - E_);
    const float* hp = h + (long)(b * N_ + src) * 1024;
#pragma unroll
    for (int j = 0; j < 4; ++j) {
      float v = a_s[(b * N_ + src) * 4 + j] + ad[j];
      v = v > 0.f ? v : 0.2f * v;
      float alpha = __expf(v - ml[j]) / dl[j];
      acc[j] += alpha * hp[j * 256 + tid];
    }
  }
#pragma unroll
  for (int j = 0; j < 4; ++j) agg[(long)bn * 1024 + j * 256 + tid] = acc[j];
}

// node = sum_n (agg + bias) = sum_n agg + N*bias
__global__ void gat_nodesum_kernel(const float* __restrict__ agg, const float* __restrict__ bias,
                                   float* __restrict__ node) {
  int idx = blockIdx.x * blockDim.x + threadIdx.x;  // 8*1024
  if (idx >= B_ * 1024) return;
  int f = idx & 1023, b = idx >> 10;
  float s = 64.f * bias[f];
  for (int n = 0; n < N_; ++n) s += agg[(long)(b * N_ + n) * 1024 + f];
  node[idx] = s;
}

__global__ void neck_kernel(const float* __restrict__ node, const float* __restrict__ W,
                            const float* __restrict__ bias, float* __restrict__ fx) {
  int idx = blockIdx.x * blockDim.x + threadIdx.x;  // 8*256
  if (idx >= B_ * NECK_) return;
  int j = idx & 255, b = idx >> 8;
  float s = bias[j];
  for (int f = 0; f < 1024; ++f) s += node[b * 1024 + f] * W[f * 256 + j];
  fx[idx] = fmaxf(s, 0.f);
}

// ---------------- decoder: 1x1 skip conv over [fx ; feats] ----------------
__global__ void skip_conv_kernel(const float* __restrict__ fx, const float* __restrict__ feat,
                                 const float* __restrict__ w, const float* __restrict__ bias,
                                 float* __restrict__ out, int C0, int S) {
  int total = B_ * C0 * S * S;
  int idx = blockIdx.x * blockDim.x + threadIdx.x;
  if (idx >= total) return;
  int x = idx % S; int r = idx / S;
  int y = r % S;   r /= S;
  int oc = r % C0; int b = r / C0;
  int cin = C0 * 513;                 // C0 * (T*N + 1)
  int sp = y * S + x;
  const float* wr = w + (long)oc * cin;
  float acc = bias[oc];
  for (int c = 0; c < C0; ++c)
    acc += wr[c] * fx[((b * C0) + c) * S * S + sp];
  const float* wr2 = wr + C0;
  for (int t = 0; t < T_; ++t)
    for (int nn = 0; nn < N_; ++nn) {
      const float* fp = feat + (long)(((b * 8 + t) * N_ + nn) * C0) * S * S + sp;
      const float* wq = wr2 + (t * N_ + nn) * C0;
      for (int cc = 0; cc < C0; ++cc) acc += wq[cc] * fp[cc * S * S];
    }
  out[idx] = acc;
}

// ConvTranspose2d k3 s2 p1 outpad1 + ReLU; w layout [Cin,Cout,3,3]
__global__ void deconv_relu_kernel(const float* __restrict__ in, const float* __restrict__ w,
                                   const float* __restrict__ bias, float* __restrict__ out,
                                   int Cin, int Cout, int S) {
  int So = 2 * S;
  int total = B_ * Cout * So * So;
  int idx = blockIdx.x * blockDim.x + threadIdx.x;
  if (idx >= total) return;
  int x = idx % So; int r = idx / So;
  int y = r % So;   r /= So;
  int oc = r % Cout; int b = r / Cout;
  float acc = bias[oc];
#pragma unroll
  for (int ky = 0; ky < 3; ++ky) {
    int ty = y + 1 - ky;
    if (ty < 0 || (ty & 1)) continue;
    int iy = ty >> 1; if (iy >= S) continue;
#pragma unroll
    for (int kx = 0; kx < 3; ++kx) {
      int tx = x + 1 - kx;
      if (tx < 0 || (tx & 1)) continue;
      int ix = tx >> 1; if (ix >= S) continue;
      for (int c = 0; c < Cin; ++c)
        acc += w[((c * Cout + oc) * 3 + ky) * 3 + kx] * in[((b * Cin + c) * S + iy) * S + ix];
    }
  }
  out[idx] = fmaxf(acc, 0.f);
}

// cls (4ch) + reg (1ch) 1x1 convs from single-channel fx [B,1,64,64]
__global__ void head_kernel(const float* __restrict__ fx, const float* __restrict__ cw,
                            const float* __restrict__ cb, const float* __restrict__ rw,
                            const float* __restrict__ rb, float* __restrict__ out) {
  int idx = blockIdx.x * blockDim.x + threadIdx.x;   // 163840
  if (idx >= 163840) return;
  if (idx < 131072) {
    int sp = idx & 4095; int k = (idx >> 12) & 3; int b = idx >> 14;
    out[idx] = fx[b * 4096 + sp] * cw[k] + cb[k];
  } else {
    int j = idx - 131072; int sp = j & 4095; int b = j >> 12;
    out[idx] = fx[b * 4096 + sp] * rw[0] + rb[0];
  }
}

// ---------------- host launcher ----------------
extern "C" void kernel_launch(void* const* d_in, const int* in_sizes, int n_in,
                              void* d_out, int out_size, void* d_ws, size_t ws_size,
                              hipStream_t stream) {
  const float* x        = (const float*)d_in[0];
  const int*   ei       = (const int*)d_in[1];
  const float* enc_w[4] = {(const float*)d_in[2], (const float*)d_in[3],
                           (const float*)d_in[4], (const float*)d_in[5]};
  const float* enc_b[4] = {(const float*)d_in[6], (const float*)d_in[7],
                           (const float*)d_in[8], (const float*)d_in[9]};
  const float* attn_in_w  = (const float*)d_in[10];
  const float* attn_in_b  = (const float*)d_in[11];
  const float* attn_out_w = (const float*)d_in[12];
  const float* attn_out_b = (const float*)d_in[13];
  const float* gat_lin_w  = (const float*)d_in[14];
  const float* gat_att_s  = (const float*)d_in[15];
  const float* gat_att_d  = (const float*)d_in[16];
  const float* gat_bias   = (const float*)d_in[17];
  const float* neck_w     = (const float*)d_in[18];
  const float* neck_b     = (const float*)d_in[19];
  const float* skip_w[4]  = {(const float*)d_in[20], (const float*)d_in[21],
                             (const float*)d_in[22], (const float*)d_in[23]};
  const float* skip_b[4]  = {(const float*)d_in[24], (const float*)d_in[25],
                             (const float*)d_in[26], (const float*)d_in[27]};
  const float* dec_w[4]   = {(const float*)d_in[28], (const float*)d_in[29],
                             (const float*)d_in[30], (const float*)d_in[31]};
  const float* dec_b[4]   = {(const float*)d_in[32], (const float*)d_in[33],
                             (const float*)d_in[34], (const float*)d_in[35]};
  const float* cls_w = (const float*)d_in[36];
  const float* cls_b = (const float*)d_in[37];
  const float* reg_w = (const float*)d_in[38];
  const float* reg_b = (const float*)d_in[39];

  float* ws = (float*)d_ws;
  size_t off = 0;
  auto alloc = [&](size_t n) { float* p = ws + off; off += n; return p; };
  float* f0   = alloc((size_t)BTN_ * 2 * 32 * 32);   // 8.4M
  float* f1   = alloc((size_t)BTN_ * 4 * 16 * 16);   // 4.2M
  float* f2   = alloc((size_t)BTN_ * 8 * 8 * 8);     // 2.1M
  float* f3   = alloc((size_t)BTN_ * 16 * 4 * 4);    // 1.0M (= x_temporal)
  float* qkv  = alloc((size_t)BTN_ * 768);
  float* oat  = alloc((size_t)BTN_ * NECK_);
  float* af   = alloc((size_t)B_ * N_ * 2048);
  float* hbuf = alloc((size_t)B_ * N_ * 1024);
  float* a_s  = alloc(2048);
  float* a_d  = alloc(2048);
  float* mxb  = alloc(2048);
  float* denb = alloc(2048);
  float* agg  = alloc((size_t)B_ * N_ * 1024);
  float* node = alloc((size_t)B_ * 1024);
  float* fxA  = alloc(32768);
  float* fxB  = alloc(32768);
  (void)ws_size; (void)in_sizes; (void)n_in; (void)out_size;

  // --- encoder (memory bound; direct conv) ---
  enc_conv_kernel<<<(BTN_ * 2 * 32 * 32 + 255) / 256, 256, 0, stream>>>(x,  enc_w[0], enc_b[0], f0, 1, 64, 2);
  enc_conv_kernel<<<(BTN_ * 4 * 16 * 16 + 255) / 256, 256, 0, stream>>>(f0, enc_w[1], enc_b[1], f1, 2, 32, 4);
  enc_conv_kernel<<<(BTN_ * 8 *  8 *  8 + 255) / 256, 256, 0, stream>>>(f1, enc_w[2], enc_b[2], f2, 4, 16, 8);
  enc_conv_kernel<<<(BTN_ * 16 * 4 *  4 + 255) / 256, 256, 0, stream>>>(f2, enc_w[3], enc_b[3], f3, 8,  8, 16);

  // --- attention (WMMA GEMMs + tiny softmax) ---
  qkv_wmma_kernel<<<64 * 4 * 48 / 8, 256, 0, stream>>>(f3, attn_in_w, attn_in_b, qkv);
  attention_kernel<<<(B_ * T_ * N_ * HEADS_ + 255) / 256, 256, 0, stream>>>(qkv, oat);
  attnout_wmma_kernel<<<64 * 4 * 16 / 8, 256, 0, stream>>>(oat, attn_out_w, attn_out_b, af);

  // --- GAT (TDM-staged A panel + WMMA) ---
  gat_wmma_kernel<<<32 * 8, 256, 0, stream>>>(af, gat_lin_w, hbuf);
  gat_attcoef_kernel<<<(2048 + 255) / 256, 256, 0, stream>>>(hbuf, gat_att_s, gat_att_d, a_s, a_d);
  gat_maxden_kernel<<<(2048 + 255) / 256, 256, 0, stream>>>(ei, a_s, a_d, mxb, denb);
  gat_agg_kernel<<<B_ * N_, 256, 0, stream>>>(ei, hbuf, a_s, a_d, mxb, denb, agg);
  gat_nodesum_kernel<<<(B_ * 1024 + 255) / 256, 256, 0, stream>>>(agg, gat_bias, node);
  neck_kernel<<<(B_ * NECK_ + 255) / 256, 256, 0, stream>>>(node, neck_w, neck_b, fxA);

  // --- decoder with dense skips ---
  const float* feats[4] = {f0, f1, f2, f3};
  int C0 = 16, S = 4;
  float* cur = fxA; float* tmp = fxB;
  for (int j = 0; j < 4; ++j) {
    int tot1 = B_ * C0 * S * S;
    skip_conv_kernel<<<(tot1 + 255) / 256, 256, 0, stream>>>(cur, feats[3 - j], skip_w[j], skip_b[j], tmp, C0, S);
    int tot2 = B_ * (C0 / 2) * (2 * S) * (2 * S);
    deconv_relu_kernel<<<(tot2 + 255) / 256, 256, 0, stream>>>(tmp, dec_w[j], dec_b[j], cur, C0, C0 / 2, S);
    C0 >>= 1; S <<= 1;
  }

  head_kernel<<<(163840 + 255) / 256, 256, 0, stream>>>(cur, cls_w, cls_b, reg_w, reg_b, (float*)d_out);
}
